// NeighborhoodAttention_21629455303332
// MI455X (gfx1250) — matrix operand order repaired and checked by reading, not compiled, not run on hardware
//
#include <hip/hip_runtime.h>

typedef __bf16 bf16_t;
typedef __bf16 v16bf __attribute__((ext_vector_type(16)));
typedef float  v8f   __attribute__((ext_vector_type(8)));
typedef float  f32x4 __attribute__((ext_vector_type(4)));
typedef unsigned int u32x4 __attribute__((ext_vector_type(4)));
typedef unsigned int u32x2 __attribute__((ext_vector_type(2)));

#define GHW 3136   // 56*56
#define NH  12
#define HD  32

union U32BF2 { unsigned u; __bf16 h[2]; };

__device__ __forceinline__ v16bf pack16(u32x4 lo, u32x4 hi) {
    union { unsigned u[8]; v16bf v; } t;
    t.u[0] = lo[0]; t.u[1] = lo[1]; t.u[2] = lo[2]; t.u[3] = lo[3];
    t.u[4] = hi[0]; t.u[5] = hi[1]; t.u[6] = hi[2]; t.u[7] = hi[3];
    return t.v;
}

__device__ __forceinline__ v8f wmma_bf16(v16bf a, v16bf b, v8f c) {
    return __builtin_amdgcn_wmma_f32_16x16x32_bf16(false, a, false, b, (short)0, c, false, false);
}

// ---------------------------------------------------------------- converts
__global__ __launch_bounds__(256) void cvt_f32_bf16(const float* __restrict__ in,
                                                    bf16_t* __restrict__ out, int n4) {
    int i = blockIdx.x * 256 + threadIdx.x;
    if (i >= n4) return;
    f32x4 v = ((const f32x4*)in)[i];
    U32BF2 a, b;
    a.h[0] = (__bf16)v[0]; a.h[1] = (__bf16)v[1];
    b.h[0] = (__bf16)v[2]; b.h[1] = (__bf16)v[3];
    ((unsigned*)out)[2 * i]     = a.u;
    ((unsigned*)out)[2 * i + 1] = b.u;
}

// in: [B][384][GHW] f32  ->  out: [B][GHW][384] bf16   (32x32 LDS tile)
__global__ __launch_bounds__(256) void transpose_cvt(const float* __restrict__ in,
                                                     bf16_t* __restrict__ out) {
    __shared__ bf16_t tile[32][33];
    const int b  = blockIdx.z;
    const int c0 = blockIdx.y * 32;
    const int p0 = blockIdx.x * 32;
    {
        const int c  = threadIdx.x >> 3;
        const int p4 = (threadIdx.x & 7) * 4;
        f32x4 v = *(const f32x4*)(in + ((size_t)b * 384 + c0 + c) * GHW + p0 + p4);
#pragma unroll
        for (int j = 0; j < 4; ++j) tile[c][p4 + j] = (bf16_t)v[j];
    }
    __syncthreads();
    {
        const int p  = threadIdx.x >> 3;
        const int c4 = (threadIdx.x & 7) * 4;
        U32BF2 a, bb;
        a.h[0]  = tile[c4][p];     a.h[1]  = tile[c4 + 1][p];
        bb.h[0] = tile[c4 + 2][p]; bb.h[1] = tile[c4 + 3][p];
        u32x2 w; w[0] = a.u; w[1] = bb.u;
        *(u32x2*)(out + ((size_t)b * GHW + p0 + p) * 384 + c0 + c4) = w;
    }
}

// ---------------------------------------------------------------- GEMM core
// 32x32 register-blocked tile per wave:
//   acc[am][an] (16x16 each) = A[m0+16*am ...] x BT[pixels p0+16*an ...]
// A: MxK row-major bf16;  BT: pixel-major bf16, element (k,p) at (b*GHW+p)*K+k.
// Per k-step: 8x global_load_b128 -> 4x v_wmma (2 A-frags x 2 B-frags).
struct Acc4 { v8f a[2][2]; };

__device__ __forceinline__ Acc4 gemm32x32(const bf16_t* __restrict__ A, int K, int m0,
                                          const bf16_t* __restrict__ BT, int b, int p0,
                                          int lane) {
    const int mi = lane & 15, hf = lane >> 4;
    const bf16_t* arow0 = A + (size_t)(m0 + mi) * K;
    const bf16_t* arow1 = arow0 + (size_t)16 * K;
    const bf16_t* brow0 = BT + ((size_t)b * GHW + p0 + mi) * K;
    const bf16_t* brow1 = brow0 + (size_t)16 * K;
    Acc4 r;
    r.a[0][0] = (v8f){}; r.a[0][1] = (v8f){};
    r.a[1][0] = (v8f){}; r.a[1][1] = (v8f){};
#pragma unroll 2
    for (int kk = 0; kk < (K >> 5); ++kk) {
        const u32x4* ap0 = (const u32x4*)(arow0 + kk * 32);
        const u32x4* ap1 = (const u32x4*)(arow1 + kk * 32);
        const u32x4* bp0 = (const u32x4*)(brow0 + kk * 32);
        const u32x4* bp1 = (const u32x4*)(brow1 + kk * 32);
        v16bf a0 = pack16(ap0[hf],     ap0[2 + hf]);     // K = 8*hf+e / 16+8*hf+e
        v16bf a1 = pack16(ap1[hf],     ap1[2 + hf]);
        v16bf b0 = pack16(bp0[hf],     bp0[2 + hf]);     // K = 8*hf+e / 16+8*hf+e
        v16bf b1 = pack16(bp1[hf],     bp1[2 + hf]);
        r.a[0][0] = wmma_bf16(a0, b0, r.a[0][0]);
        r.a[0][1] = wmma_bf16(a0, b1, r.a[0][1]);
        r.a[1][0] = wmma_bf16(a1, b0, r.a[1][0]);
        r.a[1][1] = wmma_bf16(a1, b1, r.a[1][1]);
    }
    return r;
}

// ---------------------------------------------------------------- QKV GEMM
__global__ __launch_bounds__(256) void qkv_gemm(const bf16_t* __restrict__ A,
                                                const bf16_t* __restrict__ BT,
                                                const float* __restrict__ bias,
                                                bf16_t* __restrict__ qb,
                                                bf16_t* __restrict__ kb,
                                                bf16_t* __restrict__ vb) {
    const int w = threadIdx.x >> 5, lane = threadIdx.x & 31;
    const int m0 = blockIdx.y * 32;
    const int n0 = blockIdx.x * 256 + w * 32;
    const int b = n0 / GHW, p0 = n0 % GHW;
    Acc4 acc = gemm32x32(A, 384, m0, BT, b, p0, lane);
    const int mi = lane & 15, hf = lane >> 4;
    // section (q/k/v) is uniform for the whole 32-row tile: hoist the division
    const int s  = m0 / 384;
    const int r0 = m0 % 384;
    bf16_t* dst = (s == 0) ? qb : (s == 1 ? kb : vb);
    const float scale = (s == 0) ? 0.17677669529663689f : 1.0f;  // 32^-0.5 on q
#pragma unroll
    for (int am = 0; am < 2; ++am) {
#pragma unroll
        for (int an = 0; an < 2; ++an) {
            const int p = p0 + 16 * an + mi;
#pragma unroll
            for (int v = 0; v < 8; ++v) {
                int rr = r0 + 16 * am + v + 8 * hf;         // 0..383
                int head = rr >> 5, hd = rr & 31;
                float val = (acc.a[am][an][v] + bias[m0 + 16 * am + v + 8 * hf]) * scale;
                dst[(((size_t)b * NH + head) * GHW + p) * HD + hd] = (bf16_t)val;
            }
        }
    }
}

// ---------------------------------------------------------------- proj GEMM
__global__ __launch_bounds__(256) void proj_gemm(const bf16_t* __restrict__ A,
                                                 const bf16_t* __restrict__ BT,
                                                 const float* __restrict__ bias,
                                                 float* __restrict__ out) {
    const int w = threadIdx.x >> 5, lane = threadIdx.x & 31;
    const int m0 = blockIdx.y * 32;
    const int n0 = blockIdx.x * 256 + w * 32;
    const int b = n0 / GHW, p0 = n0 % GHW;
    Acc4 acc = gemm32x32(A, 384, m0, BT, b, p0, lane);
    const int mi = lane & 15, hf = lane >> 4;
#pragma unroll
    for (int am = 0; am < 2; ++am) {
#pragma unroll
        for (int an = 0; an < 2; ++an) {
#pragma unroll
            for (int v = 0; v < 8; ++v) {
                int o = m0 + 16 * am + v + 8 * hf;
                out[((size_t)b * 384 + o) * GHW + p0 + 16 * an + mi] = acc.a[am][an][v] + bias[o];
            }
        }
    }
}

// ---------------------------------------------------------------- fused NA
// One wave per (b, head, 4x4 query tile). 10x10 clamped key grid -> 112 slots.
__global__ __launch_bounds__(128) void na_kernel(const bf16_t* __restrict__ qb,
                                                 const bf16_t* __restrict__ kb,
                                                 const bf16_t* __restrict__ vb,
                                                 const float* __restrict__ rel_bias,
                                                 bf16_t* __restrict__ attoutT) {
    __shared__ __align__(16) bf16_t sAttn[4][16 * 128];   // [query][key]
    __shared__ __align__(16) bf16_t sVT[4][HD * 128];     // [d][key]  (transposed!)
    const int lane = threadIdx.x & 31;
    const int wv = threadIdx.x >> 5;
    const int t = blockIdx.x * 4 + wv;                    // < 9408
    const int tj = t % 14, ti = (t / 14) % 14, head = (t / 196) % NH, b = t / 2352;
    const int mi = lane & 15, hf = lane >> 4;
    const size_t hb = ((size_t)b * NH + head) * (size_t)GHW * HD;
    const bf16_t* qbase = qb + hb;
    const bf16_t* kbase = kb + hb;
    const bf16_t* vbase = vb + hb;

    // ---- Q fragment (16 queries x 32 dims), pure b128 + bitcast
    const int qi = ti * 4 + (mi >> 2), qj = tj * 4 + (mi & 3);
    const u32x4* qp = (const u32x4*)(qbase + (size_t)(qi * 56 + qj) * HD);
    v16bf aq = pack16(qp[hf], qp[2 + hf]);

    // ---- stage V transposed into LDS: sVT[d][key]
    for (int s = lane; s < 128; s += 32) {
        int ks = s < 100 ? s : 99;
        int kr = ks / 10, kc = ks % 10;
        int ky = ti * 4 - 3 + kr; ky = ky < 0 ? 0 : (ky > 55 ? 55 : ky);
        int kx = tj * 4 - 3 + kc; kx = kx < 0 ? 0 : (kx > 55 ? 55 : kx);
        const u32x4* src = (const u32x4*)(vbase + (size_t)(ky * 56 + kx) * HD);
        u32x4 w0 = src[0], w1 = src[1], w2 = src[2], w3 = src[3];
        U32BF2 u;
#pragma unroll
        for (int j = 0; j < 4; ++j) {
            u.u = w0[j]; sVT[wv][(2*j    )*128 + s] = u.h[0]; sVT[wv][(2*j + 1)*128 + s] = u.h[1];
            u.u = w1[j]; sVT[wv][(8 + 2*j)*128 + s] = u.h[0]; sVT[wv][(9 + 2*j)*128 + s] = u.h[1];
            u.u = w2[j]; sVT[wv][(16+ 2*j)*128 + s] = u.h[0]; sVT[wv][(17+ 2*j)*128 + s] = u.h[1];
            u.u = w3[j]; sVT[wv][(24+ 2*j)*128 + s] = u.h[0]; sVT[wv][(25+ 2*j)*128 + s] = u.h[1];
        }
    }

    // ---- logits: 7 WMMAs, full head dim (K=32) each
    float lg[7][8];
#pragma unroll
    for (int kt = 0; kt < 7; ++kt) {
        int ks = kt * 16 + mi; int ksc = ks < 100 ? ks : 99;
        int kr = ksc / 10, kc = ksc % 10;
        int ky = ti * 4 - 3 + kr; ky = ky < 0 ? 0 : (ky > 55 ? 55 : ky);
        int kx = tj * 4 - 3 + kc; kx = kx < 0 ? 0 : (kx > 55 ? 55 : kx);
        const u32x4* kp = (const u32x4*)(kbase + (size_t)(ky * 56 + kx) * HD);
        v16bf bk = pack16(kp[hf], kp[2 + hf]);
        v8f c = {};
        c = wmma_bf16(aq, bk, c);
#pragma unroll
        for (int v = 0; v < 8; ++v) lg[kt][v] = c[v];
    }

    // ---- mask + rel_bias + softmax (keys spread across 16-lane group)
    const float* rbias = rel_bias + head * 49;
    float mx[8], sm[8];
#pragma unroll
    for (int v = 0; v < 8; ++v) {
        int m = v + 8 * hf, qr = m >> 2, qc = m & 3;
        float mv = -1e30f;
#pragma unroll
        for (int kt = 0; kt < 7; ++kt) {
            int ks = kt * 16 + mi;
            float L = -1e30f;
            if (ks < 100) {
                int kr = ks / 10, kc2 = ks % 10;
                int di = kr - qr, dj = kc2 - qc;
                if ((unsigned)di < 7u && (unsigned)dj < 7u) L = lg[kt][v] + rbias[di * 7 + dj];
            }
            lg[kt][v] = L;
            mv = fmaxf(mv, L);
        }
        mx[v] = mv;
    }
#pragma unroll
    for (int msk = 1; msk < 16; msk <<= 1)
#pragma unroll
        for (int v = 0; v < 8; ++v) mx[v] = fmaxf(mx[v], __shfl_xor(mx[v], msk, 32));
#pragma unroll
    for (int v = 0; v < 8; ++v) {
        float s = 0.f;
#pragma unroll
        for (int kt = 0; kt < 7; ++kt) { float e = __expf(lg[kt][v] - mx[v]); lg[kt][v] = e; s += e; }
        sm[v] = s;
    }
#pragma unroll
    for (int msk = 1; msk < 16; msk <<= 1)
#pragma unroll
        for (int v = 0; v < 8; ++v) sm[v] += __shfl_xor(sm[v], msk, 32);

    // ---- write attn (bf16) to LDS, pad key slots 112..127 with zero
#pragma unroll
    for (int v = 0; v < 8; ++v) {
        float inv = 1.f / sm[v];
        int m = v + 8 * hf;
#pragma unroll
        for (int kt = 0; kt < 7; ++kt)
            sAttn[wv][m * 128 + kt * 16 + mi] = (bf16_t)(lg[kt][v] * inv);
    }
#pragma unroll
    for (int r = hf; r < 16; r += 2) sAttn[wv][r * 128 + 112 + mi] = (bf16_t)0.f;
    __syncthreads();

    // ---- P*V: 4 k-steps x 2 column tiles = 8 WMMAs, all b128 LDS loads
    v8f o0 = {}, o1 = {};
    const u32x4* pr  = (const u32x4*)(sAttn[wv] + mi * 128);
    const u32x4* pv0 = (const u32x4*)(sVT[wv] + mi * 128);
    const u32x4* pv1 = (const u32x4*)(sVT[wv] + (16 + mi) * 128);
#pragma unroll
    for (int k4 = 0; k4 < 4; ++k4) {
        v16bf ap  = pack16(pr [4 * k4 + hf],     pr [4 * k4 + 2 + hf]);
        v16bf bv0 = pack16(pv0[4 * k4 + hf], pv0[4 * k4 + 2 + hf]);
        v16bf bv1 = pack16(pv1[4 * k4 + hf], pv1[4 * k4 + 2 + hf]);
        o0 = wmma_bf16(ap, bv0, o0);
        o1 = wmma_bf16(ap, bv1, o1);
    }

    // ---- write out transposed: [b][pixel][c=head*32+d]  (coalesced in mi)
#pragma unroll
    for (int v = 0; v < 8; ++v) {
        int m = v + 8 * hf;
        int pix = (ti * 4 + (m >> 2)) * 56 + tj * 4 + (m & 3);
        bf16_t* o = attoutT + ((size_t)b * GHW + pix) * 384 + head * HD;
        o[mi]      = (bf16_t)o0[v];
        o[16 + mi] = (bf16_t)o1[v];
    }
}

// ---------------------------------------------------------------- launch
extern "C" void kernel_launch(void* const* d_in, const int* in_sizes, int n_in,
                              void* d_out, int out_size, void* d_ws, size_t ws_size,
                              hipStream_t stream) {
    (void)in_sizes; (void)n_in; (void)out_size; (void)ws_size;
    const float* x        = (const float*)d_in[0];
    const float* w_qkv    = (const float*)d_in[1];
    const float* b_qkv    = (const float*)d_in[2];
    const float* w_proj   = (const float*)d_in[3];
    const float* b_proj   = (const float*)d_in[4];
    const float* rel_bias = (const float*)d_in[5];
    float* out = (float*)d_out;

    constexpr size_t N_X     = 4ull * 384 * GHW;        // 4,816,896
    constexpr size_t N_WQKV  = 1152ull * 384;
    constexpr size_t N_WPROJ = 384ull * 384;
    constexpr size_t N_HB    = 4ull * NH * GHW * HD;    // == N_X

    char* ws = (char*)d_ws;
    bf16_t* xT_bf    = (bf16_t*)ws; ws += N_X * 2;      // [b][p][c]
    bf16_t* wqkv_bf  = (bf16_t*)ws; ws += N_WQKV * 2;
    bf16_t* wproj_bf = (bf16_t*)ws; ws += N_WPROJ * 2;
    bf16_t* qbuf     = (bf16_t*)ws; ws += N_HB * 2;
    bf16_t* kbuf     = (bf16_t*)ws; ws += N_HB * 2;
    bf16_t* vbuf     = (bf16_t*)ws; ws += N_HB * 2;
    bf16_t* attbufT  = (bf16_t*)ws; ws += N_X * 2;      // [b][p][c]

    transpose_cvt<<<dim3(98, 12, 4), 256, 0, stream>>>(x, xT_bf);
    cvt_f32_bf16<<<(unsigned)((N_WQKV / 4 + 255) / 256), 256, 0, stream>>>(w_qkv, wqkv_bf, (int)(N_WQKV / 4));
    cvt_f32_bf16<<<(unsigned)((N_WPROJ / 4 + 255) / 256), 256, 0, stream>>>(w_proj, wproj_bf, (int)(N_WPROJ / 4));

    qkv_gemm<<<dim3(196, 36), 256, 0, stream>>>(wqkv_bf, xT_bf, b_qkv, qbuf, kbuf, vbuf);
    na_kernel<<<2352, 128, 0, stream>>>(qbuf, kbuf, vbuf, rel_bias, attbufT);
    proj_gemm<<<dim3(196, 12), 256, 0, stream>>>(wproj_bf, attbufT, b_proj, out);
}
